// GeometryOptimalTransport_3135326126813
// MI455X (gfx1250) — matrix operand order repaired and checked by reading, not compiled, not run on hardware
//
#include <hip/hip_runtime.h>
#include <hip/hip_bf16.h>

typedef __attribute__((ext_vector_type(16))) _Float16 v16h;
typedef __attribute__((ext_vector_type(8)))  _Float16 v8h;
typedef __attribute__((ext_vector_type(8)))  float    v8f;

#define BB 4
#define NN 4096
#define MM 4096
#define CC 256
#define EPS      (0.01f + 1e-8f)
#define THRESH2  (0.2f * 0.2f)
#define NEGINF   (-1.0e9f)
#define FARLOC   (1.0e9f)

// ---------------------------------------------------------------------------
// Row pass: u[b,m] = -logsumexp_n( logK(m,n) + v[n] ); optionally has_source.
// One thread per target row m; source locs / v staged through LDS in chunks.
// ---------------------------------------------------------------------------
__global__ __launch_bounds__(256)
void ot_row_pass(const float* __restrict__ src_locs,
                 const float* __restrict__ tgt_locs,
                 const int*   __restrict__ smask,
                 const int*   __restrict__ tmask,
                 const float* __restrict__ v,
                 float* __restrict__ u,
                 float* __restrict__ has_src,
                 int use_v, int write_has)
{
    __shared__ float sSx[512], sSy[512], sVv[512];
    const int b = blockIdx.y;
    const int m = blockIdx.x * 256 + threadIdx.x;

    const float tx = tgt_locs[(b * MM + m) * 2 + 0];
    const float ty = tgt_locs[(b * MM + m) * 2 + 1];
    const int   tval = tmask[b * MM + m];

    float mx = -3.0e38f, s = 0.0f;
    int anyv = 0;

    for (int n0 = 0; n0 < NN; n0 += 512) {
        __syncthreads();
        for (int i = threadIdx.x; i < 512; i += 256) {
            const int n  = n0 + i;
            const int sm = smask[b * NN + n];
            sSx[i] = sm ? src_locs[(b * NN + n) * 2 + 0] : FARLOC;
            sSy[i] = sm ? src_locs[(b * NN + n) * 2 + 1] : FARLOC;
            sVv[i] = use_v ? v[b * NN + n] : 0.0f;
        }
        __syncthreads();
        for (int i = 0; i < 512; ++i) {
            const float dx = tx - sSx[i], dy = ty - sSy[i];
            const float d2 = dx * dx + dy * dy;
            const bool valid = (d2 < THRESH2) && (tval != 0);
            anyv |= (int)valid;
            const float x = (valid ? (-d2 / EPS) : NEGINF) + sVv[i];
            if (x > mx) { s = s * __expf(mx - x) + 1.0f; mx = x; }
            else        { s += __expf(x - mx); }
        }
    }
    u[b * MM + m] = -(mx + __logf(s));
    if (write_has) has_src[b * MM + m] = anyv ? 1.0f : 0.0f;
}

// ---------------------------------------------------------------------------
// Col pass: v[b,n] = -logsumexp_m( logK(m,n) + u[m] ), zeroed if src invalid.
// ---------------------------------------------------------------------------
__global__ __launch_bounds__(256)
void ot_col_pass(const float* __restrict__ src_locs,
                 const float* __restrict__ tgt_locs,
                 const int*   __restrict__ smask,
                 const int*   __restrict__ tmask,
                 const float* __restrict__ u,
                 float* __restrict__ v)
{
    __shared__ float sTx[512], sTy[512], sU[512];
    const int b = blockIdx.y;
    const int n = blockIdx.x * 256 + threadIdx.x;

    const float sx = src_locs[(b * NN + n) * 2 + 0];
    const float sy = src_locs[(b * NN + n) * 2 + 1];
    const int   sval = smask[b * NN + n];

    float mx = -3.0e38f, s = 0.0f;

    for (int m0 = 0; m0 < MM; m0 += 512) {
        __syncthreads();
        for (int i = threadIdx.x; i < 512; i += 256) {
            const int m  = m0 + i;
            const int tm = tmask[b * MM + m];
            sTx[i] = tm ? tgt_locs[(b * MM + m) * 2 + 0] : FARLOC;
            sTy[i] = tm ? tgt_locs[(b * MM + m) * 2 + 1] : FARLOC;
            sU[i]  = u[b * MM + m];
        }
        __syncthreads();
        for (int i = 0; i < 512; ++i) {
            const float dx = sTx[i] - sx, dy = sTy[i] - sy;
            const float d2 = dx * dx + dy * dy;
            const bool valid = (d2 < THRESH2) && (sval != 0);
            const float x = (valid ? (-d2 / EPS) : NEGINF) + sU[i];
            if (x > mx) { s = s * __expf(mx - x) + 1.0f; mx = x; }
            else        { s += __expf(x - mx); }
        }
    }
    const float vv = -(mx + __logf(s));
    v[b * NN + n] = sval ? vv : 0.0f;
}

// ---------------------------------------------------------------------------
// Fused attn GEMM: out[b,m,c] = sum_n exp(logK + u + v) * feats[b,n,c]
// Block = 256 threads = 8 waves. Block tile: 16(M) x 256(C = all of C).
// Wave w owns columns [w*32, w*32+32) -> two 16x16 accumulators -> 2 WMMAs
// per 32-wide K step. The attn A tile (16x32) is computed cooperatively ONCE
// per block (2 exps/thread/step) and written to LDS in WMMA-fragment order so
// each lane fetches its A operand as one contiguous 32B LDS read. Feats are
// staged fp32->f16, k-contiguous (transposed), so each B fragment is one
// contiguous 32B LDS read per half-group.
// ---------------------------------------------------------------------------
__global__ __launch_bounds__(256)
void ot_attn_gemm(const float* __restrict__ feats,
                  const float* __restrict__ src_locs,
                  const float* __restrict__ tgt_locs,
                  const int*   __restrict__ smask,
                  const int*   __restrict__ tmask,
                  const float* __restrict__ u,
                  const float* __restrict__ v,
                  const float* __restrict__ has_src,
                  float* __restrict__ out)
{
    __shared__ __align__(32) _Float16 sB[256 * 32];   // [c][k] f16, 16 KB
    __shared__ __align__(32) _Float16 aLds[512];      // A tile in fragment order
    __shared__ float sTxm[16], sTym[16], sUm[16];     // masked row constants

    const int b    = blockIdx.y;
    const int m0   = blockIdx.x * 16;
    const int tid  = threadIdx.x;
    const int lane = tid & 31;
    const int wave = tid >> 5;
    const int g    = lane >> 4;    // half-group (K split in A/B layouts)
    const int lo   = lane & 15;

    // Stage masked per-row target constants once (invalid target -> FARLOC).
    if (tid < 16) {
        const int m  = m0 + tid;
        const int tm = tmask[b * MM + m];
        sTxm[tid] = tm ? tgt_locs[(b * MM + m) * 2 + 0] : FARLOC;
        sTym[tid] = tm ? tgt_locs[(b * MM + m) * 2 + 1] : FARLOC;
        sUm[tid]  = u[b * MM + m];
    }

    v8f acc0 = {};
    v8f acc1 = {};

    // Feats staging geometry: thread covers an [8k x 4c] sub-tile.
    const int c4   = tid & 63;          // float4 column index (c = c4*4)
    const int kblk = tid >> 6;          // 0..3
    const int kst  = kblk * 8;          // starting k of this thread's rows

    for (int k0 = 0; k0 < NN; k0 += 32) {
        __syncthreads();   // protect previous step's fragment reads

        // ---- stage feats [32 x 256] fp32 -> f16, transposed to [c][k] ----
        {
            const float4* fp4 =
                (const float4*)(feats + ((size_t)b * NN + k0) * CC);
            float4 f[8];
            #pragma unroll
            for (int i = 0; i < 8; ++i)
                f[i] = fp4[(size_t)(kst + i) * (CC / 4) + c4];
            #pragma unroll
            for (int j = 0; j < 4; ++j) {
                v8h hv;
                #pragma unroll
                for (int i = 0; i < 8; ++i) {
                    const float* fe = (const float*)&f[i];
                    hv[i] = (_Float16)fe[j];
                }
                *(v8h*)(&sB[(c4 * 4 + j) * 32 + kst]) = hv;
            }
        }

        // ---- prefetch next K chunk (global_prefetch_b8) ----
        if (k0 + 32 < NN)
            __builtin_prefetch(
                feats + ((size_t)b * NN + k0 + 32) * CC + tid * 32, 0, 1);

        // ---- cooperative attn tile (16x32) -> LDS in A-fragment order ----
        {
            const int e  = tid * 2;
            const int mA = e >> 5;             // row 0..15
            const float tx = sTxm[mA];
            const float ty = sTym[mA];
            const float um = sUm[mA];
            #pragma unroll
            for (int q = 0; q < 2; ++q) {
                const int k = (e & 31) + q;    // 0..31
                const int n = k0 + k;
                const int sm = smask[b * NN + n];
                const float2 sl = ((const float2*)src_locs)[b * NN + n];
                const float sx = sm ? sl.x : FARLOC;
                const float sy = sm ? sl.y : FARLOC;
                const float dx = tx - sx, dy = ty - sy;
                const float d2 = dx * dx + dy * dy;
                const float lk = (d2 < THRESH2) ? (-d2 / EPS) : NEGINF;
                const float val = __expf(lk + um + v[b * NN + n]);
                // A layout: lanes 0-15 hold K 0..15 of row m,
                //           lanes 16-31 hold K 16..31.
                const int lslot = mA + 16 * (k >> 4);
                const int elem  = k & 15;
                aLds[lslot * 16 + elem] = (_Float16)val;
            }
        }
        __syncthreads();

        // ---- fragments + 2 WMMAs per wave ----
        const v16h a = ((const v16h*)aLds)[lane];
        const v16h* sB16 = (const v16h*)sB;
        const int colA = wave * 32 + lo;
        const v16h b0 = sB16[colA * 2 + g];          // K = g*16 .. g*16+15
        const v16h b1 = sB16[(colA + 16) * 2 + g];

        acc0 = __builtin_amdgcn_wmma_f32_16x16x32_f16(
            false, a, false, b0, (short)0, acc0, false, false);
        acc1 = __builtin_amdgcn_wmma_f32_16x16x32_f16(
            false, a, false, b1, (short)0, acc1, false, false);
    }

    // C/D layout: VGPR r -> row m0 + r + 8*g, column = lane&15.
    #pragma unroll
    for (int r = 0; r < 8; ++r) {
        const int mr = m0 + r + 8 * g;
        const float scale =
            (tmask[b * MM + mr] ? 1.0f : 0.0f) * has_src[b * MM + mr];
        const size_t base = ((size_t)b * MM + mr) * CC;
        out[base + wave * 32 + lo]      = acc0[r] * scale;
        out[base + wave * 32 + 16 + lo] = acc1[r] * scale;
    }
}

// ---------------------------------------------------------------------------
extern "C" void kernel_launch(void* const* d_in, const int* in_sizes, int n_in,
                              void* d_out, int out_size, void* d_ws, size_t ws_size,
                              hipStream_t stream) {
    const float* feats = (const float*)d_in[0];  // [B,N,C] f32
    const float* slocs = (const float*)d_in[1];  // [B,N,2] f32
    const float* tlocs = (const float*)d_in[2];  // [B,M,2] f32
    const int*   smask = (const int*)d_in[3];    // [B,N]
    const int*   tmask = (const int*)d_in[4];    // [B,M]
    float* out = (float*)d_out;                  // [B,M,C] f32

    float* ws  = (float*)d_ws;                   // 3*B*4096 floats = 192 KB
    float* u   = ws;                             // [B,M]
    float* v   = ws + BB * MM;                   // [B,N]
    float* has = ws + BB * MM + BB * NN;         // [B,M]

    dim3 blk(256);
    dim3 rg(MM / 256, BB);
    dim3 cg(NN / 256, BB);

    // 3 Sinkhorn iterations (u then v each); first row pass also records
    // has_source and uses implicit v = 0.
    ot_row_pass<<<rg, blk, 0, stream>>>(slocs, tlocs, smask, tmask, v, u, has, 0, 1);
    ot_col_pass<<<cg, blk, 0, stream>>>(slocs, tlocs, smask, tmask, u, v);
    ot_row_pass<<<rg, blk, 0, stream>>>(slocs, tlocs, smask, tmask, v, u, has, 1, 0);
    ot_col_pass<<<cg, blk, 0, stream>>>(slocs, tlocs, smask, tmask, u, v);
    ot_row_pass<<<rg, blk, 0, stream>>>(slocs, tlocs, smask, tmask, v, u, has, 1, 0);
    ot_col_pass<<<cg, blk, 0, stream>>>(slocs, tlocs, smask, tmask, u, v);

    dim3 gg(MM / 16, BB);
    ot_attn_gemm<<<gg, blk, 0, stream>>>(feats, slocs, tlocs, smask, tmask,
                                         u, v, has, out);
}
